// GMatcher_23691039604786
// MI455X (gfx1250) — matrix operand order repaired and checked by reading, not compiled, not run on hardware
//
#include <hip/hip_runtime.h>
#include <math.h>

// ---------------------------------------------------------------------------
// Types for CDNA5 WMMA (wave32, V_WMMA_F32_16X16X32_F16)
// ---------------------------------------------------------------------------
typedef __attribute__((ext_vector_type(16))) _Float16 v16h;
typedef __attribute__((ext_vector_type(8)))  _Float16 v8h;
typedef __attribute__((ext_vector_type(8)))  float    v8f;

// Problem constants (match reference)
#define BB  2
#define NN  1024
#define DD  256
#define HH  4
#define HDIM 64
#define LL  18
#define NP  1025   // N+1

// ---------------------------------------------------------------------------
// Generic batched GEMM:  C[z] = alpha * A[z] * B[z] + bias
//   A element (m,k) = OPA ? A[k*lda + m] : A[m*lda + k]
//   B element (k,n) = OPB ? B[n*ldb + k] : B[k*ldb + n]
//   C element (m,n) at C[m*ldc + n]
//   per-z base offsets: (z/inner)*Out + (z%inner)*In  for A,B,C
//
// Block: 256 threads = 8 wave32 waves; block tile 128(M) x 128(N);
// wave tile 32x64 = 2x4 WMMA frags; K stepped by 32 through double-buffered
// LDS (f32 -> f16 conversion on stage-in), one barrier per K step.
//
// Out-of-range M/N rows are handled by *address clamping only* (garbage rows
// only reach C entries masked at the store); only K-overflow zero-fills (GK).
// Strided staging uses a compile-time stride (LDT) where known so all 16
// loads become immediate-offset loads off one base address (single clause).
// ---------------------------------------------------------------------------
struct GemmP {
    const float* A; const float* B; const float* bias; float* C;
    int M, Nn, K, lda, ldb, ldc, inner;
    float alpha;
    long long aOut, aIn, bOut, bIn, cOut, cIn;
};

#define LDS_STRIDE 40   // 32 + 8 halves pad: 80B rows -> 16B aligned, conflict-free

// Stage 16 k-consecutive elements of one (already clamped) row into LDS.
// CONTIG: elements contiguous at src[row*ld + k]; else strided src[k*ld + row].
// GK:     zero-fill k >= K (branch-free select); else K assumed multiple of 32.
// LDT:    compile-time stride for the strided path (0 = runtime ld).
template<bool CONTIG, bool GK, int LDT>
__device__ __forceinline__ void stage16(_Float16* dst, const float* src,
                                        long long row, int ld, int k0, int K) {
    v8h t0 = {}, t1 = {};
    if (CONTIG && !GK) {
        const float4* pv = reinterpret_cast<const float4*>(src + row * (long long)ld + k0);
        float4 a = pv[0], b = pv[1], c = pv[2], d = pv[3];
        __builtin_prefetch(src + row * (long long)ld + k0 + 64, 0, 1);  // speculative, next-next tile
        t0[0]=(_Float16)a.x; t0[1]=(_Float16)a.y; t0[2]=(_Float16)a.z; t0[3]=(_Float16)a.w;
        t0[4]=(_Float16)b.x; t0[5]=(_Float16)b.y; t0[6]=(_Float16)b.z; t0[7]=(_Float16)b.w;
        t1[0]=(_Float16)c.x; t1[1]=(_Float16)c.y; t1[2]=(_Float16)c.z; t1[3]=(_Float16)c.w;
        t1[4]=(_Float16)d.x; t1[5]=(_Float16)d.y; t1[6]=(_Float16)d.z; t1[7]=(_Float16)d.w;
    } else if (!CONTIG && !GK && LDT > 0) {
        // strided with compile-time stride: one base, 16 immediate-offset loads
        const float* base = src + (long long)k0 * LDT + row;
        float f[16];
        #pragma unroll
        for (int e = 0; e < 16; ++e) f[e] = base[(long long)e * LDT];
        #pragma unroll
        for (int e = 0; e < 8; ++e) { t0[e] = (_Float16)f[e]; t1[e] = (_Float16)f[e + 8]; }
    } else {
        float f[16];
        #pragma unroll
        for (int e = 0; e < 16; ++e) {
            const int gk = k0 + e;
            long long idx = CONTIG ? row * (long long)ld + gk
                                   : (long long)gk * ld + row;
            if (GK) {
                const bool ok = gk < K;
                idx = ok ? idx : 0;          // clamped address, no exec divergence
                f[e] = src[idx];
                f[e] = ok ? f[e] : 0.f;
            } else {
                f[e] = src[idx];
            }
        }
        #pragma unroll
        for (int e = 0; e < 8; ++e) { t0[e] = (_Float16)f[e]; t1[e] = (_Float16)f[e + 8]; }
    }
    *reinterpret_cast<v8h*>(dst)     = t0;
    *reinterpret_cast<v8h*>(dst + 8) = t1;
}

template<bool OPA, bool OPB, bool GK, int LDA_C, int LDB_C>
__global__ __launch_bounds__(256) void gemm_wmma_kernel(GemmP p) {
    // double-buffered tiles: A [128 m][32 k], B transposed to [128 n][32 k]
    __shared__ __align__(16) _Float16 As[2][128 * LDS_STRIDE];
    __shared__ __align__(16) _Float16 Bs[2][128 * LDS_STRIDE];

    const int tid  = threadIdx.x;
    const int lane = tid & 31;
    const int wave = tid >> 5;
    const int z    = blockIdx.z;

    const float* A  = p.A + (long long)(z / p.inner) * p.aOut + (long long)(z % p.inner) * p.aIn;
    const float* Bm = p.B + (long long)(z / p.inner) * p.bOut + (long long)(z % p.inner) * p.bIn;
    float*       C  = p.C + (long long)(z / p.inner) * p.cOut + (long long)(z % p.inner) * p.cIn;

    const int bm0 = blockIdx.y * 128;
    const int bn0 = blockIdx.x * 128;
    const int wm  = (wave & 3) * 32;     // wave M offset in block tile
    const int wn  = (wave >> 2) * 64;    // wave N offset in block tile

    // staging assignment: each thread stages one row-half (16 k) of A and of B
    const int trow = tid >> 1;           // 0..127
    const int tkh  = (tid & 1) * 16;     // 0 or 16

    // clamped global rows: out-of-range rows stage garbage that is masked at store
    const int gmA = (bm0 + trow < p.M)  ? bm0 + trow : p.M - 1;
    const int gnB = (bn0 + trow < p.Nn) ? bn0 + trow : p.Nn - 1;

    const int mr = lane & 15;            // fragment row/col within 16
    const int hi = lane >> 4;            // lane-half select

    v8f acc[2][4] = {};

    // ---- prologue: stage first K tile into buffer 0 ----
    stage16<!OPA, GK, LDA_C>(&As[0][trow * LDS_STRIDE + tkh], A,  gmA, p.lda, tkh, p.K);
    stage16< OPB, GK, LDB_C>(&Bs[0][trow * LDS_STRIDE + tkh], Bm, gnB, p.ldb, tkh, p.K);

    int cur = 0;
    for (int kt = 0; kt < p.K; kt += 32) {
        __syncthreads();                 // staged tile `cur` visible to all waves
        if (kt + 32 < p.K) {             // overlap next-tile staging with compute
            stage16<!OPA, GK, LDA_C>(&As[cur ^ 1][trow * LDS_STRIDE + tkh], A,  gmA, p.lda, kt + 32 + tkh, p.K);
            stage16< OPB, GK, LDB_C>(&Bs[cur ^ 1][trow * LDS_STRIDE + tkh], Bm, gnB, p.ldb, kt + 32 + tkh, p.K);
        }

        // ---- fragment loads per ISA 7.12.2 layouts (ds_load_b128) ----
        v16h af[2], bf[4];
        #pragma unroll
        for (int i = 0; i < 2; ++i) {
            // A 16x32: lane m=mr, VGPR0-3: K=8*hi+0..7, VGPR4-7: K=16+8*hi+0..7
            const _Float16* pa = &As[cur][(wm + 16 * i + mr) * LDS_STRIDE + 8 * hi];
            v8h lo  = *reinterpret_cast<const v8h*>(pa);
            v8h hi8 = *reinterpret_cast<const v8h*>(pa + 16);
            af[i] = __builtin_shufflevector(lo, hi8, 0,1,2,3,4,5,6,7,8,9,10,11,12,13,14,15);
        }
        #pragma unroll
        for (int j = 0; j < 4; ++j) {
            // B 32x16: lane n=mr, VGPR0-3: K=8*hi+0..7, VGPR4-7: K=16+8*hi+0..7
            const _Float16* pb = &Bs[cur][(wn + 16 * j + mr) * LDS_STRIDE + 8 * hi];
            v8h lo  = *reinterpret_cast<const v8h*>(pb);
            v8h hi8 = *reinterpret_cast<const v8h*>(pb + 16);
            bf[j] = __builtin_shufflevector(lo, hi8, 0,1,2,3,4,5,6,7,8,9,10,11,12,13,14,15);
        }
        #pragma unroll
        for (int i = 0; i < 2; ++i)
            #pragma unroll
            for (int j = 0; j < 4; ++j)
                acc[i][j] = __builtin_amdgcn_wmma_f32_16x16x32_f16(
                    false, af[i], false, bf[j], (short)0, acc[i][j], false, false);
        cur ^= 1;
    }

    // ---- epilogue: alpha, bias, store (C/D layout: VGPR r -> row r+8*hi) ----
    const bool fullTile = (bm0 + 128 <= p.M) && (bn0 + 128 <= p.Nn);  // uniform per block
    if (fullTile) {
        #pragma unroll
        for (int i = 0; i < 2; ++i) {
            #pragma unroll
            for (int j = 0; j < 4; ++j) {
                const int col = bn0 + wn + 16 * j + mr;
                #pragma unroll
                for (int r = 0; r < 8; ++r) {
                    const int row = bm0 + wm + 16 * i + r + 8 * hi;
                    float v = acc[i][j][r] * p.alpha;
                    if (p.bias) v += p.bias[row];
                    C[(long long)row * p.ldc + col] = v;
                }
            }
        }
    } else {
        #pragma unroll
        for (int i = 0; i < 2; ++i) {
            #pragma unroll
            for (int j = 0; j < 4; ++j) {
                const int col = bn0 + wn + 16 * j + mr;
                if (col < p.Nn) {
                    #pragma unroll
                    for (int r = 0; r < 8; ++r) {
                        const int row = bm0 + wm + 16 * i + r + 8 * hi;
                        if (row < p.M) {
                            float v = acc[i][j][r] * p.alpha;
                            if (p.bias) v += p.bias[row];
                            C[(long long)row * p.ldc + col] = v;
                        }
                    }
                }
            }
        }
    }
}

// ---------------------------------------------------------------------------
// BatchNorm1d (training stats over batch*positions per channel) + ReLU.
// x layout [B, C, N]; one block per channel. Single read pass: each thread
// holds its 8 values (B*N/256) in registers.
// ---------------------------------------------------------------------------
__global__ void bn_relu_kernel(float* x, const float* g, const float* bt, int C, int relu) {
    const int c = blockIdx.x, tid = threadIdx.x;
    __shared__ float s_sum[256];
    __shared__ float s_sq[256];
    float vals[8];
    long long offs[8];
    float sum = 0.f, sq = 0.f;
    #pragma unroll
    for (int q = 0; q < 8; ++q) {
        const int idx = tid + 256 * q;           // 0 .. B*N-1 (2048)
        const int b = idx >> 10, n = idx & (NN - 1);
        offs[q] = ((long long)b * C + c) * NN + n;
        vals[q] = x[offs[q]];
        sum += vals[q]; sq += vals[q] * vals[q];
    }
    s_sum[tid] = sum; s_sq[tid] = sq; __syncthreads();
    for (int s = 128; s > 0; s >>= 1) {
        if (tid < s) { s_sum[tid] += s_sum[tid + s]; s_sq[tid] += s_sq[tid + s]; }
        __syncthreads();
    }
    const float mean = s_sum[0] / (float)(BB * NN);
    const float var  = s_sq[0] / (float)(BB * NN) - mean * mean;
    const float inv  = rsqrtf(var + 1e-5f);
    const float gg = g[c], bb = bt[c];
    #pragma unroll
    for (int q = 0; q < 8; ++q) {
        float v = (vals[q] - mean) * inv * gg + bb;
        if (relu) v = fmaxf(v, 0.f);
        x[offs[q]] = v;
    }
}

// ---------------------------------------------------------------------------
// Row softmax over 1024 columns (in place), one block per row.
// Register-resident: one read pass + one write pass.
// ---------------------------------------------------------------------------
__global__ void softmax_rows_kernel(float* s) {
    float* r = s + (long long)blockIdx.x * NN;
    const int tid = threadIdx.x;
    __shared__ float red[256];
    float vals[4];
    float mx = -3.0e38f;
    #pragma unroll
    for (int q = 0; q < 4; ++q) { vals[q] = r[tid + 256 * q]; mx = fmaxf(mx, vals[q]); }
    red[tid] = mx; __syncthreads();
    for (int st = 128; st > 0; st >>= 1) {
        if (tid < st) red[tid] = fmaxf(red[tid], red[tid + st]);
        __syncthreads();
    }
    mx = red[0]; __syncthreads();
    float se = 0.f;
    #pragma unroll
    for (int q = 0; q < 4; ++q) { vals[q] = expf(vals[q] - mx); se += vals[q]; }
    red[tid] = se; __syncthreads();
    for (int st = 128; st > 0; st >>= 1) {
        if (tid < st) red[tid] += red[tid + st];
        __syncthreads();
    }
    const float inv = 1.f / red[0];
    #pragma unroll
    for (int q = 0; q < 4; ++q) r[tid + 256 * q] = vals[q] * inv;
}

// ---------------------------------------------------------------------------
// Elementwise helpers
// ---------------------------------------------------------------------------
__global__ void add2_kernel(float* dst, const float* a, const float* b, long long n) {
    const long long i = (long long)blockIdx.x * blockDim.x + threadIdx.x;
    if (i < n) dst[i] = a[i] + b[i];
}
__global__ void add_ip_kernel(float* dst, const float* a, long long n) {
    const long long i = (long long)blockIdx.x * blockDim.x + threadIdx.x;
    if (i < n) dst[i] += a[i];
}
__global__ void zero_kernel(float* p, long long n) {
    const long long i = (long long)blockIdx.x * blockDim.x + threadIdx.x;
    if (i < n) p[i] = 0.f;
}
// h[b, c, n] = c<D ? x[b,c,n] : m[b,c-D,n]   (h is [B, 2D, N])
__global__ void concat_kernel(float* h, const float* x, const float* m) {
    const long long total = (long long)BB * 2 * DD * NN;
    const long long idx = (long long)blockIdx.x * blockDim.x + threadIdx.x;
    if (idx >= total) return;
    const int n = (int)(idx % NN);
    const long long rem = idx / NN;
    const int c = (int)(rem % (2 * DD));
    const int b = (int)(rem / (2 * DD));
    h[idx] = (c < DD) ? x[((long long)b * DD + c) * NN + n]
                      : m[((long long)b * DD + (c - DD)) * NN + n];
}

// ---------------------------------------------------------------------------
// Sinkhorn (log optimal transport)
// ---------------------------------------------------------------------------
__global__ void fill_bins_kernel(float* coup, const float* alpha_p) {
    const float a = *alpha_p;
    const int idx = blockIdx.x * blockDim.x + threadIdx.x;
    if (idx >= BB * NP) return;
    const int b = idx / NP, i = idx % NP;
    coup[((long long)b * NP + i) * NP + NN] = a;   // last column
    coup[((long long)b * NP + NN) * NP + i] = a;   // last row
}

// u[b,i] = log_mu[i] - lse_j( coup[b,i,j] + v[b,j] ).  Single read pass:
// each thread holds 5 values (1025 = 4*256 + 1); invalid slots get -3e38
// whose exp underflows to zero in the sum.
__global__ void sink_row_kernel(const float* coup, const float* vv, float* uu) {
    const int row = blockIdx.x;                 // b*NP + i
    const int b = row / NP, i = row % NP;
    const float* cr = coup + ((long long)b * NP + i) * NP;
    const float* vb = vv + (long long)b * NP;
    const int tid = threadIdx.x;
    __shared__ float red[256];
    float vals[5];
    float mx = -3.0e38f;
    #pragma unroll
    for (int q = 0; q < 5; ++q) {
        const int j = tid + 256 * q;
        vals[q] = (j < NP) ? (cr[j] + vb[j]) : -3.0e38f;
        mx = fmaxf(mx, vals[q]);
    }
    red[tid] = mx; __syncthreads();
    for (int st = 128; st > 0; st >>= 1) { if (tid < st) red[tid] = fmaxf(red[tid], red[tid + st]); __syncthreads(); }
    mx = red[0]; __syncthreads();
    float se = 0.f;
    #pragma unroll
    for (int q = 0; q < 5; ++q) se += expf(vals[q] - mx);   // invalid -> exp(-inf) = 0
    red[tid] = se; __syncthreads();
    for (int st = 128; st > 0; st >>= 1) { if (tid < st) red[tid] += red[tid + st]; __syncthreads(); }
    if (tid == 0) {
        const float lse  = mx + logf(red[0]);
        const float norm = -logf((float)(2 * NN));
        const float lmu  = (i < NN) ? norm : (logf((float)NN) + norm);
        uu[row] = lmu - lse;
    }
}

// v[b,j] = log_nu[j] - lse_i( coup[b,i,j] + u[b,i] )
__global__ void sink_col_kernel(const float* coup, const float* uu, float* vv) {
    const int cidx = blockIdx.x;                // b*NP + j
    const int b = cidx / NP, j = cidx % NP;
    const float* cb = coup + (long long)b * NP * NP + j;
    const float* ub = uu + (long long)b * NP;
    const int tid = threadIdx.x;
    __shared__ float red[256];
    float vals[5];
    float mx = -3.0e38f;
    #pragma unroll
    for (int q = 0; q < 5; ++q) {
        const int i = tid + 256 * q;
        vals[q] = (i < NP) ? (cb[(long long)i * NP] + ub[i]) : -3.0e38f;
        mx = fmaxf(mx, vals[q]);
    }
    red[tid] = mx; __syncthreads();
    for (int st = 128; st > 0; st >>= 1) { if (tid < st) red[tid] = fmaxf(red[tid], red[tid + st]); __syncthreads(); }
    mx = red[0]; __syncthreads();
    float se = 0.f;
    #pragma unroll
    for (int q = 0; q < 5; ++q) se += expf(vals[q] - mx);
    red[tid] = se; __syncthreads();
    for (int st = 128; st > 0; st >>= 1) { if (tid < st) red[tid] += red[tid + st]; __syncthreads(); }
    if (tid == 0) {
        const float lse  = mx + logf(red[0]);
        const float norm = -logf((float)(2 * NN));
        const float lnu  = (j < NN) ? norm : (logf((float)NN) + norm);
        vv[cidx] = lnu - lse;
    }
}

// out = couplings + u[:,:,None] + v[:,None,:] - norm
__global__ void final_out_kernel(float* out, const float* coup, const float* uu, const float* vv) {
    const long long total = (long long)BB * NP * NP;
    const long long idx = (long long)blockIdx.x * blockDim.x + threadIdx.x;
    if (idx >= total) return;
    const int j = (int)(idx % NP);
    const long long rem = idx / NP;
    const int i = (int)(rem % NP);
    const int b = (int)(rem / NP);
    out[idx] = coup[idx] + uu[(long long)b * NP + i] + vv[(long long)b * NP + j]
             + logf((float)(2 * NN));   // minus norm, norm = -log(m+n)
}

// ---------------------------------------------------------------------------
// Host orchestration
// ---------------------------------------------------------------------------
typedef void (*GemmFn)(GemmP);

static GemmFn pick_gemm(int opA, int opB, bool gk, int lda, int ldb) {
    if (opA) {   // A strided, B strided (scores / final score GEMMs)
        if (lda == 4096 && ldb == 4096) return (GemmFn)gemm_wmma_kernel<true, false, false, 4096, 4096>;
        if (lda == 1024 && ldb == 1024) return (GemmFn)gemm_wmma_kernel<true, false, false, 1024, 1024>;
        return (GemmFn)gemm_wmma_kernel<true, false, false, 0, 0>;
    }
    if (opB) {   // both contiguous (attn out / kenc L0)
        return gk ? (GemmFn)gemm_wmma_kernel<false, true, true,  0, 0>
                  : (GemmFn)gemm_wmma_kernel<false, true, false, 0, 0>;
    }
    // A contiguous, B strided (all projections); ldb == 1024 everywhere
    if (ldb == 1024) return (GemmFn)gemm_wmma_kernel<false, false, false, 0, 1024>;
    return (GemmFn)gemm_wmma_kernel<false, false, false, 0, 0>;
}

extern "C" void kernel_launch(void* const* d_in, const int* in_sizes, int n_in,
                              void* d_out, int out_size, void* d_ws, size_t ws_size,
                              hipStream_t stream) {
    (void)in_sizes; (void)n_in; (void)out_size; (void)ws_size;
    // Input order = setup_inputs() dict insertion order, tuples/dicts flattened in order:
    const float* kpts[2] = { (const float*)d_in[0], (const float*)d_in[1] };
    const float* desc[2] = { (const float*)d_in[2], (const float*)d_in[3] };
    const float* keW[4]  = { (const float*)d_in[4], (const float*)d_in[5],
                             (const float*)d_in[6], (const float*)d_in[7] };
    const float* keB[4]  = { (const float*)d_in[8], (const float*)d_in[9],
                             (const float*)d_in[10], (const float*)d_in[11] };
    const float* keG[3]  = { (const float*)d_in[12], (const float*)d_in[13], (const float*)d_in[14] };
    const float* keBt[3] = { (const float*)d_in[15], (const float*)d_in[16], (const float*)d_in[17] };
    const float* Wqkv = (const float*)d_in[18];
    const float* bqkv = (const float*)d_in[19];
    const float* Wm   = (const float*)d_in[20];
    const float* bm   = (const float*)d_in[21];
    const float* W1   = (const float*)d_in[22];
    const float* b1   = (const float*)d_in[23];
    const float* g1   = (const float*)d_in[24];
    const float* be1  = (const float*)d_in[25];
    const float* W2   = (const float*)d_in[26];
    const float* b2   = (const float*)d_in[27];
    const float* fpW  = (const float*)d_in[28];
    const float* fpb  = (const float*)d_in[29];
    const float* bin  = (const float*)d_in[30];
    float* out = (float*)d_out;

    // workspace layout (floats)
    const long long SZ = (long long)BB * DD * NN;   // 524288
    float* w = (float*)d_ws;
    float* d0v = w;            w += SZ;
    float* d1v = w;            w += SZ;
    float* qb  = w;            w += SZ;
    float* kb  = w;            w += SZ;
    float* vb  = w;            w += SZ;
    float* ob  = w;            w += SZ;
    float* mb  = w;            w += SZ;
    float* hb  = w;            w += 2 * SZ;
    float* h2b = w;            w += 2 * SZ;
    float* dl0 = w;            w += SZ;
    float* dl1 = w;            w += SZ;
    float* sc  = w;            w += (long long)BB * HH * NN * NN;  // 8388608
    float* coup= w;            w += (long long)BB * NP * NP;       // 2101250
    float* uu  = w;            w += BB * NP;
    float* vv  = w;            w += BB * NP;

    auto gemm = [&](const float* A, const float* Bm_, const float* bias, float* C,
                    int M, int Nn, int K, int lda, int ldb, int ldc,
                    int opA, int opB, float alpha, int batches, int inner,
                    long long aOut, long long aIn, long long bOut, long long bIn,
                    long long cOut, long long cIn) {
        GemmP p{A, Bm_, bias, C, M, Nn, K, lda, ldb, ldc, inner, alpha,
                aOut, aIn, bOut, bIn, cOut, cIn};
        const bool gk = (K % 32 != 0);
        dim3 grid((Nn + 127) / 128, (M + 127) / 128, batches);
        hipLaunchKernelGGL(pick_gemm(opA, opB, gk, lda, ldb), grid, dim3(256), 0, stream, p);
    };

    // ---- Keypoint encoder: MLP [2,32,64,128,256], BN+ReLU between ----
    for (int s = 0; s < 2; ++s) {
        float* dst = s ? d1v : d0v;
        // L0: W[32,2] x kpts^T ([B,2,N] via opB on [B,N,2], ldb=2)
        gemm(keW[0], kpts[s], keB[0], qb, 32, NN, 2, 2, 2, NN, 0, 1, 1.f,
             BB, 1, 0, 0, (long long)NN * 2, 0, (long long)32 * NN, 0);
        hipLaunchKernelGGL(bn_relu_kernel, dim3(32), dim3(256), 0, stream, qb, keG[0], keBt[0], 32, 1);
        gemm(keW[1], qb, keB[1], kb, 64, NN, 32, 32, NN, NN, 0, 0, 1.f,
             BB, 1, 0, 0, (long long)32 * NN, 0, (long long)64 * NN, 0);
        hipLaunchKernelGGL(bn_relu_kernel, dim3(64), dim3(256), 0, stream, kb, keG[1], keBt[1], 64, 1);
        gemm(keW[2], kb, keB[2], qb, 128, NN, 64, 64, NN, NN, 0, 0, 1.f,
             BB, 1, 0, 0, (long long)64 * NN, 0, (long long)128 * NN, 0);
        hipLaunchKernelGGL(bn_relu_kernel, dim3(128), dim3(256), 0, stream, qb, keG[2], keBt[2], 128, 1);
        gemm(keW[3], qb, keB[3], kb, 256, NN, 128, 128, NN, NN, 0, 0, 1.f,
             BB, 1, 0, 0, (long long)128 * NN, 0, (long long)256 * NN, 0);
        // d = desc + kenc
        hipLaunchKernelGGL(add2_kernel, dim3((unsigned)((SZ + 255) / 256)), dim3(256), 0, stream,
                           dst, desc[s], kb, SZ);
    }

    // ---- GNN: 18 layers of self/cross attention ----
    const long long DN = (long long)DD * NN;       // 262144 (per-batch activation stride)
    for (int i = 0; i < LL; ++i) {
        const int cross = i & 1;
        const float* Wq = Wqkv + ((long long)i * 3 + 0) * DD * DD;
        const float* Wk = Wqkv + ((long long)i * 3 + 1) * DD * DD;
        const float* Wv = Wqkv + ((long long)i * 3 + 2) * DD * DD;
        const float* bq = bqkv + ((long long)i * 3 + 0) * DD;
        const float* bk = bqkv + ((long long)i * 3 + 1) * DD;
        const float* bv = bqkv + ((long long)i * 3 + 2) * DD;
        const float* Wm_i = Wm + (long long)i * DD * DD;
        const float* bm_i = bm + (long long)i * DD;
        const float* W1_i = W1 + (long long)i * 4 * DD * DD;
        const float* b1_i = b1 + (long long)i * 2 * DD;
        const float* g1_i = g1 + (long long)i * 2 * DD;
        const float* be_i = be1 + (long long)i * 2 * DD;
        const float* W2_i = W2 + (long long)i * 2 * DD * DD;
        const float* b2_i = b2 + (long long)i * DD;

        for (int s = 0; s < 2; ++s) {
            const float* x   = s ? d1v : d0v;
            const float* src = cross ? (s ? d0v : d1v) : x;
            float* dl = s ? dl1 : dl0;

            // QKV projections: [256,256] x [B,256,N]
            gemm(Wq, x,   bq, qb, DD, NN, DD, DD, NN, NN, 0, 0, 1.f, BB, 1, 0, 0, DN, 0, DN, 0);
            gemm(Wk, src, bk, kb, DD, NN, DD, DD, NN, NN, 0, 0, 1.f, BB, 1, 0, 0, DN, 0, DN, 0);
            gemm(Wv, src, bv, vb, DD, NN, DD, DD, NN, NN, 0, 0, 1.f, BB, 1, 0, 0, DN, 0, DN, 0);

            // scores[b,h,n,m] = (1/8) sum_d q[b,d*H+h,n] k[b,d*H+h,m]
            // z = b*H + h; head-slice channel stride: lda=ldb=H*N=4096
            gemm(qb, kb, nullptr, sc, NN, NN, HDIM, HH * NN, HH * NN, NN, 1, 0,
                 0.125f, BB * HH, HH, DN, NN, DN, NN,
                 (long long)HH * NN * NN, (long long)NN * NN);
            hipLaunchKernelGGL(softmax_rows_kernel, dim3(BB * HH * NN), dim3(256), 0, stream, sc);

            // o[b,d*H+h,n] = sum_m p[b,h,n,m] v[b,d*H+h,m]   (V * P^T)
            gemm(vb, sc, nullptr, ob, HDIM, NN, NN, HH * NN, NN, HH * NN, 0, 1,
                 1.f, BB * HH, HH, DN, NN, (long long)HH * NN * NN, (long long)NN * NN,
                 DN, NN);

            // merge projection
            gemm(Wm_i, ob, bm_i, mb, DD, NN, DD, DD, NN, NN, 0, 0, 1.f, BB, 1, 0, 0, DN, 0, DN, 0);

            // h = concat([x, msg]); MLP: W1 (512x512) -> BN+ReLU -> W2 (256x512)
            hipLaunchKernelGGL(concat_kernel, dim3((unsigned)((2 * SZ + 255) / 256)), dim3(256), 0, stream,
                               hb, x, mb);
            gemm(W1_i, hb, b1_i, h2b, 2 * DD, NN, 2 * DD, 2 * DD, NN, NN, 0, 0, 1.f,
                 BB, 1, 0, 0, 2 * DN, 0, 2 * DN, 0);
            hipLaunchKernelGGL(bn_relu_kernel, dim3(2 * DD), dim3(256), 0, stream, h2b, g1_i, be_i, 2 * DD, 1);
            gemm(W2_i, h2b, b2_i, dl, DD, NN, 2 * DD, 2 * DD, NN, NN, 0, 0, 1.f,
                 BB, 1, 0, 0, 2 * DN, 0, DN, 0);
        }
        // residual updates (deltas computed from pre-update states)
        hipLaunchKernelGGL(add_ip_kernel, dim3((unsigned)((SZ + 255) / 256)), dim3(256), 0, stream, d0v, dl0, SZ);
        hipLaunchKernelGGL(add_ip_kernel, dim3((unsigned)((SZ + 255) / 256)), dim3(256), 0, stream, d1v, dl1, SZ);
    }

    // ---- final projections + score GEMM into couplings ----
    gemm(fpW, d0v, fpb, qb, DD, NN, DD, DD, NN, NN, 0, 0, 1.f, BB, 1, 0, 0, DN, 0, DN, 0);
    gemm(fpW, d1v, fpb, kb, DD, NN, DD, DD, NN, NN, 0, 0, 1.f, BB, 1, 0, 0, DN, 0, DN, 0);
    // scores[b,n,m] = (1/16) sum_d m0[b,d,n] m1[b,d,m]; write into couplings[:, :N, :N] (ldc=NP)
    gemm(qb, kb, nullptr, coup, NN, NN, DD, NN, NN, NP, 1, 0, 1.f / 16.f,
         BB, 1, DN, 0, DN, 0, (long long)NP * NP, 0);
    hipLaunchKernelGGL(fill_bins_kernel, dim3((BB * NP + 255) / 256), dim3(256), 0, stream, coup, bin);

    // ---- Sinkhorn: 100 iterations of u/v log-sum-exp updates ----
    hipLaunchKernelGGL(zero_kernel, dim3((BB * NP + 255) / 256), dim3(256), 0, stream, uu, (long long)BB * NP);
    hipLaunchKernelGGL(zero_kernel, dim3((BB * NP + 255) / 256), dim3(256), 0, stream, vv, (long long)BB * NP);
    for (int it = 0; it < 100; ++it) {
        hipLaunchKernelGGL(sink_row_kernel, dim3(BB * NP), dim3(256), 0, stream, coup, vv, uu);
        hipLaunchKernelGGL(sink_col_kernel, dim3(BB * NP), dim3(256), 0, stream, coup, uu, vv);
    }
    const long long TOT = (long long)BB * NP * NP;
    hipLaunchKernelGGL(final_out_kernel, dim3((unsigned)((TOT + 255) / 256)), dim3(256), 0, stream,
                       out, coup, uu, vv);
}
